// ConnectionInteractionSensingModule_67525475827944
// MI455X (gfx1250) — matrix operand order repaired and checked by reading, not compiled, not run on hardware
//
#include <hip/hip_runtime.h>
#include <hip/hip_bf16.h>
#include <math.h>

// ---------------------------------------------------------------------------
// Sizes (fixed by the reference)
#define BZ   64
#define NN   16384          // B*N nodes
#define DD   512
#define HH   8
#define HDIM 64

typedef __attribute__((ext_vector_type(16))) __bf16 v16bf;
typedef __attribute__((ext_vector_type(8)))  float  v8f;

union BF16x16 { v16bf v; unsigned int u[8]; };

__device__ __forceinline__ unsigned short f2bf(float f) {
  unsigned int u = __float_as_uint(f);
  u += 0x7FFFu + ((u >> 16) & 1u);   // round-to-nearest-even
  return (unsigned short)(u >> 16);
}

// 16-bit A-matrix 16x32 fragment K index (ISA 7.12.2)
__device__ __forceinline__ int kidxA(int v, int kh) {
  return (v < 4 ? 2 * v : 16 + 2 * (v - 4)) + 8 * kh;
}
// 16-bit B-matrix 32x16 fragment K index
__device__ __forceinline__ int kidxB(int v, int kh) {
  return (v < 4 ? 2 * v : 16 + 2 * (v - 4)) + 8 * kh;
}

// LDS offset of a __shared__ pointer (low 32 bits of the generic address)
__device__ __forceinline__ unsigned lds_off(const void* p) {
  return (unsigned)(size_t)p;
}

// Async copy of 32 contiguous bytes global->LDS (2x B128), ASYNCcnt-tracked.
__device__ __forceinline__ void async_copy32(unsigned lds, const void* gaddr) {
  unsigned long long ga = (unsigned long long)(size_t)gaddr;
  asm volatile(
      "global_load_async_to_lds_b128 %0, %1, off\n\t"
      "global_load_async_to_lds_b128 %0, %1, off offset:16"
      :: "v"(lds), "v"(ga) : "memory");
}
__device__ __forceinline__ void wait_async0() {
  asm volatile("s_wait_asynccnt 0" ::: "memory");
}

// ---------------------------------------------------------------------------
__global__ void cast_f32_bf16(const float* __restrict__ src,
                              unsigned short* __restrict__ dst, int n) {
  int i = blockIdx.x * blockDim.x + threadIdx.x;
  if (i < n) dst[i] = f2bf(src[i]);
}

__global__ void init_kernel(float* __restrict__ deg, float* __restrict__ gsum,
                            float* __restrict__ gsq) {
  int i = blockIdx.x * blockDim.x + threadIdx.x;
  if (i < NN) deg[i] = 1.0f;                 // self-loop weight
  if (i < DD) { gsum[i] = 0.0f; gsq[i] = 0.0f; }
}

// ---------------------------------------------------------------------------
// C[M=16384, N=512] = A(bf16) x W(bf16), optional f32 and/or bf16 output,
// optional two column biases. 128x128 block tile, 8 waves of 32x64.
// Double-buffered LDS; A tiles via async-to-LDS DMA, B tiles register-staged
// (transposed on ds_store_b16).
#define ASTR  40            // halfwords; 80B rows: 16B aligned, conflict-free
#define ATILE (128 * ASTR)  // halfwords per tile buffer
__global__ __launch_bounds__(256) void gemm_bf16(
    const unsigned short* __restrict__ A,   // [NN,512]
    const unsigned short* __restrict__ W,   // [512,512]
    float* __restrict__ outF,
    unsigned short* __restrict__ outB,
    const float* __restrict__ bias0,
    const float* __restrict__ bias1) {
  __shared__ __align__(16) unsigned short lA[2 * ATILE];
  __shared__ __align__(16) unsigned short lB[2 * ATILE];  // transposed [n][k]

  const int tid  = threadIdx.x;
  const int lane = tid & 31;
  const int wid  = tid >> 5;
  const int lm   = lane & 15;
  const int kh   = (lane >> 4) & 1;
  const int row0 = blockIdx.x * 128;
  const int col0 = blockIdx.y * 128;
  const int wm   = (wid & 3) * 32;
  const int wn   = (wid >> 2) * 64;

  v8f c[2][4];
#pragma unroll
  for (int mt = 0; mt < 2; ++mt)
#pragma unroll
    for (int nt = 0; nt < 4; ++nt) c[mt][nt] = (v8f)(0.0f);

  const int ar = tid >> 1;
  const int ac = (tid & 1) * 16;
  const int bk = tid >> 3;
  const int bc = (tid & 7) * 16;

  const unsigned short* gA = A + (size_t)(row0 + ar) * 512 + ac;  // + k0
  const unsigned short* gB = W + (size_t)bk * 512 + col0 + bc;    // + k0*512
  const unsigned ldsAoff0 = lds_off(&lA[ar * ASTR + ac]);

  uint4 br0, br1;  // B tile staging registers

  // ---- prologue: tile 0 ----
  async_copy32(ldsAoff0, gA);
  br0 = *(const uint4*)gB;
  br1 = *(const uint4*)(gB + 8);
  {
    unsigned short h[16];
    *(uint4*)&h[0] = br0; *(uint4*)&h[8] = br1;
#pragma unroll
    for (int j = 0; j < 16; ++j) lB[(bc + j) * ASTR + bk] = h[j];
  }
  wait_async0();
  __syncthreads();

  for (int i = 0; i < 16; ++i) {
    const int cur = i & 1, nxt = cur ^ 1;
    const int k0n = (i + 1) * 32;
    if (i < 15) {  // prefetch next tile: A async DMA, B into registers
      async_copy32(ldsAoff0 + (unsigned)(nxt * ATILE * 2), gA + k0n);
      br0 = *(const uint4*)(gB + (size_t)k0n * 512);
      br1 = *(const uint4*)(gB + (size_t)k0n * 512 + 8);
    }
    // ---- compute from current buffers ----
    const unsigned short* lAc = &lA[cur * ATILE];
    const unsigned short* lBc = &lB[cur * ATILE];
    BF16x16 af[2];
#pragma unroll
    for (int mt = 0; mt < 2; ++mt) {
      int r = wm + mt * 16 + lm;
#pragma unroll
      for (int v = 0; v < 8; ++v)
        af[mt].u[v] = *(const unsigned int*)&lAc[r * ASTR + kidxA(v, kh)];
    }
#pragma unroll
    for (int nt = 0; nt < 4; ++nt) {
      BF16x16 bf;
      int n = wn + nt * 16 + lm;
#pragma unroll
      for (int v = 0; v < 8; ++v)
        bf.u[v] = *(const unsigned int*)&lBc[n * ASTR + kidxB(v, kh)];
#pragma unroll
      for (int mt = 0; mt < 2; ++mt)
        c[mt][nt] = __builtin_amdgcn_wmma_f32_16x16x32_bf16(
            false, af[mt].v, false, bf.v, (short)0, c[mt][nt], false, false);
    }
    if (i < 15) {  // store staged B into next buffer (transposed)
      unsigned short h[16];
      *(uint4*)&h[0] = br0; *(uint4*)&h[8] = br1;
      unsigned short* base = &lB[nxt * ATILE];
#pragma unroll
      for (int j = 0; j < 16; ++j) base[(bc + j) * ASTR + bk] = h[j];
    }
    wait_async0();
    __syncthreads();
  }

#pragma unroll
  for (int mt = 0; mt < 2; ++mt)
#pragma unroll
    for (int nt = 0; nt < 4; ++nt)
#pragma unroll
      for (int r = 0; r < 8; ++r) {
        int row = row0 + wm + mt * 16 + r + 8 * kh;
        int col = col0 + wn + nt * 16 + lm;
        float v = c[mt][nt][r];
        if (bias0) v += bias0[col];
        if (bias1) v += bias1[col];
        if (outF) outF[(size_t)row * 512 + col] = v;
        if (outB) outB[(size_t)row * 512 + col] = f2bf(v);
      }
}

// ---------------------------------------------------------------------------
// Flash attention per (batch, head): Q 256x64 in registers per wave,
// K (async DMA) / V^T chunks (64 keys) staged in LDS, online softmax.
#define QSTR 72  // 144B rows: 16B aligned, conflict-free for row-indexed reads
__global__ __launch_bounds__(256) void attn_kernel(
    const unsigned short* __restrict__ Q, const unsigned short* __restrict__ K,
    const unsigned short* __restrict__ V, float* __restrict__ acc) {
  __shared__ __align__(16) unsigned short Kc[64 * QSTR];
  __shared__ __align__(16) unsigned short VTc[64 * QSTR];       // [hd][key]
  __shared__ __align__(16) unsigned short Ps[8 * 32 * QSTR];    // per-wave P

  const int b = blockIdx.x >> 3;
  const int h = blockIdx.x & 7;
  const int tid = threadIdx.x;
  const int lane = tid & 31;
  const int w = tid >> 5;
  const int lm = lane & 15;
  const int kh = (lane >> 4) & 1;
  const size_t hoff = (size_t)h * HDIM;
  const int m0 = w * 32;

  BF16x16 qf[2][2];
#pragma unroll
  for (int mt = 0; mt < 2; ++mt) {
    int row = b * 256 + m0 + mt * 16 + lm;
    const unsigned short* qp = Q + (size_t)row * 512 + hoff;
#pragma unroll
    for (int ks = 0; ks < 2; ++ks)
#pragma unroll
      for (int v = 0; v < 8; ++v)
        qf[mt][ks].u[v] = *(const unsigned int*)(qp + ks * 32 + kidxA(v, kh));
  }

  v8f o[2][4];
  float mi[2][8], li[2][8];
#pragma unroll
  for (int mt = 0; mt < 2; ++mt) {
#pragma unroll
    for (int nt = 0; nt < 4; ++nt) o[mt][nt] = (v8f)(0.0f);
#pragma unroll
    for (int r = 0; r < 8; ++r) { mi[mt][r] = -3.0e38f; li[mt][r] = 0.0f; }
  }

  const int kr = tid >> 2;
  const int seg = (tid & 3) * 16;
  const unsigned ldsKoff = lds_off(&Kc[kr * QSTR + seg]);

  for (int kc = 0; kc < 4; ++kc) {
    __syncthreads();
    // K chunk row-major via async DMA to LDS
    async_copy32(ldsKoff,
                 K + (size_t)(b * 256 + kc * 64 + kr) * 512 + hoff + seg);
    {  // V chunk transposed to [hd][key]
      const unsigned short* gp =
          V + (size_t)(b * 256 + kc * 64 + kr) * 512 + hoff + seg;
      uint4 d0 = *(const uint4*)gp;
      uint4 d1 = *(const uint4*)(gp + 8);
      unsigned short hb[16];
      *(uint4*)&hb[0] = d0; *(uint4*)&hb[8] = d1;
#pragma unroll
      for (int j = 0; j < 16; ++j) VTc[(seg + j) * QSTR + kr] = hb[j];
    }
    wait_async0();
    __syncthreads();

    v8f s[2][4];
#pragma unroll
    for (int mt = 0; mt < 2; ++mt)
#pragma unroll
      for (int nt = 0; nt < 4; ++nt) s[mt][nt] = (v8f)(0.0f);
#pragma unroll
    for (int ks = 0; ks < 2; ++ks)
#pragma unroll
      for (int nt = 0; nt < 4; ++nt) {
        BF16x16 bfK;
        int key = nt * 16 + lm;
#pragma unroll
        for (int v = 0; v < 8; ++v)
          bfK.u[v] =
              *(const unsigned int*)&Kc[key * QSTR + ks * 32 + kidxB(v, kh)];
#pragma unroll
        for (int mt = 0; mt < 2; ++mt)
          s[mt][nt] = __builtin_amdgcn_wmma_f32_16x16x32_bf16(
              false, qf[mt][ks].v, false, bfK.v, (short)0, s[mt][nt], false,
              false);
      }

    // online softmax
#pragma unroll
    for (int mt = 0; mt < 2; ++mt)
#pragma unroll
      for (int r = 0; r < 8; ++r) {
        float rmax = -3.0e38f;
#pragma unroll
        for (int nt = 0; nt < 4; ++nt) {
          s[mt][nt][r] *= 0.125f;  // 1/sqrt(64)
          rmax = fmaxf(rmax, s[mt][nt][r]);
        }
#pragma unroll
        for (int off = 8; off >= 1; off >>= 1)
          rmax = fmaxf(rmax, __shfl_xor(rmax, off, 32));
        float mnew = fmaxf(mi[mt][r], rmax);
        float corr = __expf(mi[mt][r] - mnew);
        float rsum = 0.0f;
#pragma unroll
        for (int nt = 0; nt < 4; ++nt) {
          float p = __expf(s[mt][nt][r] - mnew);
          s[mt][nt][r] = p;
          rsum += p;
        }
#pragma unroll
        for (int off = 8; off >= 1; off >>= 1)
          rsum += __shfl_xor(rsum, off, 32);
        li[mt][r] = li[mt][r] * corr + rsum;
        mi[mt][r] = mnew;
#pragma unroll
        for (int nt = 0; nt < 4; ++nt) o[mt][nt][r] *= corr;
      }

    // stage P (bf16) in per-wave LDS, then O += P * V
    unsigned short* pw = &Ps[w * 32 * QSTR];
#pragma unroll
    for (int mt = 0; mt < 2; ++mt)
#pragma unroll
      for (int nt = 0; nt < 4; ++nt)
#pragma unroll
        for (int r = 0; r < 8; ++r)
          pw[(mt * 16 + r + 8 * kh) * QSTR + nt * 16 + lm] = f2bf(s[mt][nt][r]);
    asm volatile("s_wait_dscnt 0" ::: "memory");

#pragma unroll
    for (int ks = 0; ks < 2; ++ks) {
      BF16x16 pf[2];
#pragma unroll
      for (int mt = 0; mt < 2; ++mt) {
        int prow = mt * 16 + lm;
#pragma unroll
        for (int v = 0; v < 8; ++v)
          pf[mt].u[v] =
              *(const unsigned int*)&pw[prow * QSTR + ks * 32 + kidxA(v, kh)];
      }
#pragma unroll
      for (int nt = 0; nt < 4; ++nt) {
        BF16x16 bfV;
        int hd = nt * 16 + lm;
#pragma unroll
        for (int v = 0; v < 8; ++v)
          bfV.u[v] =
              *(const unsigned int*)&VTc[hd * QSTR + ks * 32 + kidxB(v, kh)];
#pragma unroll
        for (int mt = 0; mt < 2; ++mt)
          o[mt][nt] = __builtin_amdgcn_wmma_f32_16x16x32_bf16(
              false, pf[mt].v, false, bfV.v, (short)0, o[mt][nt], false, false);
      }
    }
  }

#pragma unroll
  for (int mt = 0; mt < 2; ++mt)
#pragma unroll
    for (int r = 0; r < 8; ++r) {
      int node = b * 256 + m0 + mt * 16 + r + 8 * kh;
      float inv = 1.0f / li[mt][r];
#pragma unroll
      for (int nt = 0; nt < 4; ++nt) {
        size_t idx = (size_t)node * 512 + hoff + nt * 16 + lm;
        acc[idx] += o[mt][nt][r] * inv;  // exclusive region, serialized stream
      }
    }
}

// ---------------------------------------------------------------------------
__global__ void deg_kernel(const int* __restrict__ dst,
                           const float* __restrict__ w, float* __restrict__ deg,
                           int E) {
  int e = blockIdx.x * blockDim.x + threadIdx.x;
  if (e < E) atomicAdd(&deg[dst[e]], w[e]);
}

__global__ void dinv_kernel(float* __restrict__ deg) {
  int i = blockIdx.x * blockDim.x + threadIdx.x;
  if (i < NN) { float d = deg[i]; deg[i] = d > 0.0f ? rsqrtf(d) : 0.0f; }
}

__global__ void selfloop_kernel(const float* __restrict__ xl,
                                const float* __restrict__ dinv,
                                float* __restrict__ acc) {
  int i = blockIdx.x * blockDim.x + threadIdx.x;  // over NN*512
  int node = i >> 9;
  float di = dinv[node];
  acc[i] += di * di * xl[i];
}

__global__ __launch_bounds__(128) void edge_kernel(
    const int* __restrict__ src, const int* __restrict__ dst,
    const float* __restrict__ w, const float* __restrict__ dinv,
    const float* __restrict__ xl, float* __restrict__ acc) {
  int e = blockIdx.x;
  int s = src[e], d = dst[e];
  float nw = dinv[s] * dinv[d] * w[e];
  const float4* xr = (const float4*)(xl + (size_t)s * 512);
  float* ar = acc + (size_t)d * 512;
  int t = threadIdx.x;
  float4 x4 = xr[t];
  atomicAdd(&ar[t * 4 + 0], nw * x4.x);
  atomicAdd(&ar[t * 4 + 1], nw * x4.y);
  atomicAdd(&ar[t * 4 + 2], nw * x4.z);
  atomicAdd(&ar[t * 4 + 3], nw * x4.w);
}

// ---------------------------------------------------------------------------
__global__ __launch_bounds__(256) void bn_partial(const float* __restrict__ acc,
                                                  float* __restrict__ gsum,
                                                  float* __restrict__ gsq) {
  int t = threadIdx.x;
  int r0 = blockIdx.x * 256;
  float s0 = 0, q0 = 0, s1 = 0, q1 = 0;
  for (int r = 0; r < 256; ++r) {
    float a = acc[(size_t)(r0 + r) * 512 + t];
    float b = acc[(size_t)(r0 + r) * 512 + t + 256];
    s0 += a; q0 += a * a; s1 += b; q1 += b * b;
  }
  atomicAdd(&gsum[t], s0);       atomicAdd(&gsq[t], q0);
  atomicAdd(&gsum[t + 256], s1); atomicAdd(&gsq[t + 256], q1);
}

__global__ void finalize_kernel(float* __restrict__ out,
                                const float* __restrict__ gsum,
                                const float* __restrict__ gsq,
                                const float* __restrict__ gamma,
                                const float* __restrict__ beta) {
  int i = blockIdx.x * blockDim.x + threadIdx.x;
  int c = i & 511;
  float mean = gsum[c] * (1.0f / 16384.0f);
  float var = gsq[c] * (1.0f / 16384.0f) - mean * mean;
  float v = (out[i] - mean) * rsqrtf(var + 1e-5f) * gamma[c] + beta[c];
  out[i] = 0.5f * v * (1.0f + erff(v * 0.70710678118654752f));  // exact GELU
}

// ---------------------------------------------------------------------------
extern "C" void kernel_launch(void* const* d_in, const int* in_sizes, int n_in,
                              void* d_out, int out_size, void* d_ws,
                              size_t ws_size, hipStream_t stream) {
  const float* x    = (const float*)d_in[0];
  const int* eidx   = (const int*)d_in[1];
  const float* ew   = (const float*)d_in[2];
  const float* Wq   = (const float*)d_in[4];  const float* bq = (const float*)d_in[5];
  const float* Wk   = (const float*)d_in[6];  const float* bk = (const float*)d_in[7];
  const float* Wv   = (const float*)d_in[8];  const float* bv = (const float*)d_in[9];
  const float* Wsk  = (const float*)d_in[10]; const float* bs = (const float*)d_in[11];
  const float* Wg   = (const float*)d_in[12]; const float* bg = (const float*)d_in[13];
  const float* gam  = (const float*)d_in[14]; const float* bet = (const float*)d_in[15];
  const int E = in_sizes[2];
  float* acc = (float*)d_out;  // fused accumulator == final output buffer

  char* ws = (char*)d_ws;
  size_t off = 0;
  auto alloc = [&](size_t bytes) {
    void* p = ws + off;
    off = (off + bytes + 255) & ~(size_t)255;
    return p;
  };
  unsigned short* xb  = (unsigned short*)alloc((size_t)NN * DD * 2);
  unsigned short* wqb = (unsigned short*)alloc((size_t)DD * DD * 2);
  unsigned short* wkb = (unsigned short*)alloc((size_t)DD * DD * 2);
  unsigned short* wvb = (unsigned short*)alloc((size_t)DD * DD * 2);
  unsigned short* wsb = (unsigned short*)alloc((size_t)DD * DD * 2);
  unsigned short* wgb = (unsigned short*)alloc((size_t)DD * DD * 2);
  unsigned short* qb  = (unsigned short*)alloc((size_t)NN * DD * 2);
  unsigned short* kb  = (unsigned short*)alloc((size_t)NN * DD * 2);
  unsigned short* vb  = (unsigned short*)alloc((size_t)NN * DD * 2);
  float* xl   = (float*)alloc((size_t)NN * DD * 4);
  float* deg  = (float*)alloc((size_t)NN * 4);
  float* gsum = (float*)alloc((size_t)DD * 4);
  float* gsq  = (float*)alloc((size_t)DD * 4);

  const int n1 = NN * DD, n2 = DD * DD;
  cast_f32_bf16<<<(n1 + 255) / 256, 256, 0, stream>>>(x, xb, n1);
  cast_f32_bf16<<<(n2 + 255) / 256, 256, 0, stream>>>(Wq, wqb, n2);
  cast_f32_bf16<<<(n2 + 255) / 256, 256, 0, stream>>>(Wk, wkb, n2);
  cast_f32_bf16<<<(n2 + 255) / 256, 256, 0, stream>>>(Wv, wvb, n2);
  cast_f32_bf16<<<(n2 + 255) / 256, 256, 0, stream>>>(Wsk, wsb, n2);
  cast_f32_bf16<<<(n2 + 255) / 256, 256, 0, stream>>>(Wg, wgb, n2);
  init_kernel<<<64, 256, 0, stream>>>(deg, gsum, gsq);

  dim3 gg(128, 4);
  gemm_bf16<<<gg, 256, 0, stream>>>(xb, wsb, acc, nullptr, bs, bg);      // skip
  gemm_bf16<<<gg, 256, 0, stream>>>(xb, wqb, nullptr, qb, bq, nullptr);  // Q
  gemm_bf16<<<gg, 256, 0, stream>>>(xb, wkb, nullptr, kb, bk, nullptr);  // K
  gemm_bf16<<<gg, 256, 0, stream>>>(xb, wvb, nullptr, vb, bv, nullptr);  // V
  gemm_bf16<<<gg, 256, 0, stream>>>(xb, wgb, xl, nullptr, nullptr, nullptr);

  deg_kernel<<<(E + 255) / 256, 256, 0, stream>>>(eidx + E, ew, deg, E);
  dinv_kernel<<<NN / 256, 256, 0, stream>>>(deg);
  selfloop_kernel<<<n1 / 256, 256, 0, stream>>>(xl, deg, acc);
  edge_kernel<<<E, 128, 0, stream>>>(eidx, eidx + E, ew, deg, xl, acc);

  attn_kernel<<<BZ * HH, 256, 0, stream>>>(qb, kb, vb, acc);

  bn_partial<<<NN / 256, 256, 0, stream>>>(acc, gsum, gsq);
  finalize_kernel<<<n1 / 256, 256, 0, stream>>>(acc, gsum, gsq, gam, bet);
}
